// AffNetR_77077483094221
// MI455X (gfx1250) — matrix operand order repaired and checked by reading, not compiled, not run on hardware
//
#include <hip/hip_runtime.h>
#include <hip/hip_bf16.h>
#include <math.h>

// Problem dims (fixed by the reference setup)
#define E_DIM 128
#define U_DIM 8192
#define I_DIM 8192
#define H_DIM 4
#define EPSV  1e-6f

typedef __attribute__((ext_vector_type(16))) __bf16 v16bf;
typedef __attribute__((ext_vector_type(8)))  float  v8f;

union Frag {
    v16bf v;
    uint4 q[2];
};

__device__ __forceinline__ unsigned short f2bf_rne(float f) {
    unsigned u = __float_as_uint(f);
    u += 0x7FFFu + ((u >> 16) & 1u);
    return (unsigned short)(u >> 16);
}
__device__ __forceinline__ float bf2f(unsigned short h) {
    return __uint_as_float(((unsigned)h) << 16);
}

// One 128-thread block per row of length E_DIM.
// Computes v = x / (||row|| + eps), splits v into bf16 hi + bf16 lo planes.
__global__ void norm_split_kernel(const float* __restrict__ src,
                                  unsigned short* __restrict__ hi,
                                  unsigned short* __restrict__ lo) {
    __shared__ float red[E_DIM];
    const int row = blockIdx.x;
    const int e   = threadIdx.x;
    const size_t idx = (size_t)row * E_DIM + e;

    float x = src[idx];
    red[e] = x * x;
    __syncthreads();
    #pragma unroll
    for (int s = E_DIM / 2; s > 0; s >>= 1) {
        if (e < s) red[e] += red[e + s];
        __syncthreads();
    }
    float inv = 1.0f / (sqrtf(red[0]) + EPSV);
    float v = x * inv;
    unsigned short h = f2bf_rne(v);
    float r = v - bf2f(h);
    hi[idx] = h;
    lo[idx] = f2bf_rne(r);
}

// Tiled GEMM + cross-head max + epilogue.
// Grid: (I/128, U/128). Block: 256 threads = 8 waves.
// Wave tile: 64 (U) x 32 (I) = 4x2 fragments of 16x16.
// A = Zn rows (M=u, K=e), B = Xn rows as columns (N=i, K=e): pure NT layout.
// All tile base pointers are wave-uniform (SGPR saddr); per-lane addresses are
// loop-invariant 32-bit offsets -> global_load_b128 in GVS form, no per-load
// VALU 64-bit address math (and no WMMA->VALU WAR hazard NOPs).
__global__ __launch_bounds__(256) void affmax_gemm_kernel(
    const unsigned short* __restrict__ Xhi, const unsigned short* __restrict__ Xlo,
    const unsigned short* __restrict__ Zhi, const unsigned short* __restrict__ Zlo,
    const float* __restrict__ betaPtr, float* __restrict__ out)
{
    const int tid   = threadIdx.x;
    const int lane  = tid & 31;
    const int wid   = __builtin_amdgcn_readfirstlane(tid >> 5);  // wave-uniform
    const int l16   = lane & 15;
    const int lHalf = lane >> 4;

    const int uWave = blockIdx.y * 128 + (wid & 1) * 64;   // scalar
    const int iWave = blockIdx.x * 128 + (wid >> 1) * 32;  // scalar

    // Loop-invariant per-lane element offsets into the tile panels.
    unsigned offA[4], offB[2];
    #pragma unroll
    for (int m = 0; m < 4; ++m)
        offA[m] = (unsigned)((m * 16 + l16) * E_DIM + lHalf * 8);
    #pragma unroll
    for (int n = 0; n < 2; ++n)
        offB[n] = (unsigned)((n * 16 + l16) * E_DIM + lHalf * 8);

    v8f maxv[4][2];
    #pragma unroll
    for (int m = 0; m < 4; ++m)
        #pragma unroll
        for (int n = 0; n < 2; ++n)
            #pragma unroll
            for (int j = 0; j < 8; ++j)
                maxv[m][n][j] = -3.402823466e38f;

    const unsigned short* __restrict__ pXhiT = Xhi + (size_t)iWave * E_DIM; // scalar base
    const unsigned short* __restrict__ pXloT = Xlo + (size_t)iWave * E_DIM;

    #pragma unroll 1
    for (int h = 0; h < H_DIM; ++h) {
        const unsigned short* __restrict__ ZhT =
            Zhi + (size_t)h * U_DIM * E_DIM + (size_t)uWave * E_DIM;  // scalar base
        const unsigned short* __restrict__ ZlT =
            Zlo + (size_t)h * U_DIM * E_DIM + (size_t)uWave * E_DIM;

        v8f acc[4][2];
        #pragma unroll
        for (int m = 0; m < 4; ++m)
            #pragma unroll
            for (int n = 0; n < 2; ++n)
                #pragma unroll
                for (int j = 0; j < 8; ++j)
                    acc[m][n][j] = 0.0f;

        #pragma unroll 1
        for (int k0 = 0; k0 < E_DIM; k0 += 32) {
            const unsigned short* pBh = pXhiT + k0;   // scalar + k (SGPR add)
            const unsigned short* pBl = pXloT + k0;
            const unsigned short* pAh = ZhT + k0;
            const unsigned short* pAl = ZlT + k0;

            // B fragments: lane holds column (i); V0-3: K=k0+lHalf*8..+7,
            // V4-7: K=k0+16+lHalf*8..+7.
            Frag bh[2], bl[2];
            #pragma unroll
            for (int n = 0; n < 2; ++n) {
                bh[n].q[0] = *(const uint4*)(pBh + offB[n]);
                bh[n].q[1] = *(const uint4*)(pBh + offB[n] + 16);
                bl[n].q[0] = *(const uint4*)(pBl + offB[n]);
                bl[n].q[1] = *(const uint4*)(pBl + offB[n] + 16);
            }
            #pragma unroll
            for (int m = 0; m < 4; ++m) {
                // A fragment: lane holds row (u); V0-3: K=k0+lHalf*8..+7,
                // V4-7: K=k0+16+lHalf*8..+7.
                Frag ah, al;
                ah.q[0] = *(const uint4*)(pAh + offA[m]);
                ah.q[1] = *(const uint4*)(pAh + offA[m] + 16);
                al.q[0] = *(const uint4*)(pAl + offA[m]);
                al.q[1] = *(const uint4*)(pAl + offA[m] + 16);

                #pragma unroll
                for (int n = 0; n < 2; ++n) {
                    // split product: hi*hi + hi*lo + lo*hi (drop lo*lo)
                    acc[m][n] = __builtin_amdgcn_wmma_f32_16x16x32_bf16(
                        false, ah.v, false, bh[n].v, (short)0, acc[m][n], false, false);
                    acc[m][n] = __builtin_amdgcn_wmma_f32_16x16x32_bf16(
                        false, ah.v, false, bl[n].v, (short)0, acc[m][n], false, false);
                    acc[m][n] = __builtin_amdgcn_wmma_f32_16x16x32_bf16(
                        false, al.v, false, bh[n].v, (short)0, acc[m][n], false, false);
                }
            }
        }

        #pragma unroll
        for (int m = 0; m < 4; ++m)
            #pragma unroll
            for (int n = 0; n < 2; ++n)
                #pragma unroll
                for (int j = 0; j < 8; ++j)
                    maxv[m][n][j] = fmaxf(maxv[m][n][j], acc[m][n][j]);
    }

    const float beta = *betaPtr;

    // C/D layout: VGPR r, lanes 0-15 -> row M=r, lanes 16-31 -> row M=r+8.
    // Streaming output: non-temporal stores (don't rinse 256MB through L2).
    float* outBase = out + (size_t)uWave * I_DIM + iWave;   // scalar base
    #pragma unroll
    for (int m = 0; m < 4; ++m) {
        #pragma unroll
        for (int n = 0; n < 2; ++n) {
            #pragma unroll
            for (int r = 0; r < 8; ++r) {
                const unsigned off =
                    (unsigned)((m * 16 + lHalf * 8 + r) * I_DIM + n * 16 + l16);
                float v = (maxv[m][n][r] + 1.0f) * 0.5f;
                float o = (beta == 1.0f) ? v : powf(v, beta);
                __builtin_nontemporal_store(o, outBase + off);
            }
        }
    }
}

extern "C" void kernel_launch(void* const* d_in, const int* in_sizes, int n_in,
                              void* d_out, int out_size, void* d_ws, size_t ws_size,
                              hipStream_t stream) {
    const float* X    = (const float*)d_in[0];   // [I, E]
    const float* Z    = (const float*)d_in[1];   // [H, U, E]
    const float* beta = (const float*)d_in[2];   // scalar
    float* out        = (float*)d_out;           // [U, I]

    // Workspace: bf16 hi/lo planes of normalized X and Z (20 MB total)
    unsigned short* Xhi = (unsigned short*)d_ws;
    unsigned short* Xlo = Xhi + (size_t)I_DIM * E_DIM;
    unsigned short* Zhi = Xlo + (size_t)I_DIM * E_DIM;
    unsigned short* Zlo = Zhi + (size_t)H_DIM * U_DIM * E_DIM;

    norm_split_kernel<<<I_DIM,          E_DIM, 0, stream>>>(X, Xhi, Xlo);
    norm_split_kernel<<<H_DIM * U_DIM,  E_DIM, 0, stream>>>(Z, Zhi, Zlo);

    dim3 grid(I_DIM / 128, U_DIM / 128);
    affmax_gemm_kernel<<<grid, 256, 0, stream>>>(Xhi, Xlo, Zhi, Zlo, beta, out);
}
